// _GCNEncoder_50242527428955
// MI455X (gfx1250) — matrix operand order repaired and checked by reading, not compiled, not run on hardware
//
#include <hip/hip_runtime.h>
#include <hip/hip_bf16.h>

typedef __attribute__((ext_vector_type(16))) _Float16 v16h;
typedef __attribute__((ext_vector_type(8)))  float    v8f;

// ---------------------------------------------------------------------------
// Degree / normalization
// ---------------------------------------------------------------------------
__global__ void deg_init_kernel(float* deg, int N) {
    int i = blockIdx.x * blockDim.x + threadIdx.x;
    if (i < N) deg[i] = 1.0f;  // self-loop weight
}

__global__ void deg_scatter_kernel(float* deg, const int* ei, const float* ew, int E) {
    int e = blockIdx.x * blockDim.x + threadIdx.x;
    if (e < E) atomicAdd(&deg[ei[(size_t)E + e]], ew[e]);  // dst row
}

__global__ void deg_rsqrt_kernel(float* deg, int N) {
    int i = blockIdx.x * blockDim.x + threadIdx.x;
    if (i < N) {
        float v = deg[i];
        deg[i] = (v > 0.0f) ? rsqrtf(v) : 0.0f;
    }
}

// ---------------------------------------------------------------------------
// GEMM: Y[M x Ncols] = X[M x K] * W[K x Ncols], one wave per 16x16 tile.
// f16 inputs, f32 accumulate via v_wmma_f32_16x16x32_f16.  K is a compile-time
// constant so the K-loop fully unrolls (loads pipelined across the WMMAs).
// X is row-major with leading dimension K.
// ---------------------------------------------------------------------------
template <int K>
__device__ __forceinline__ void gemm_tile_body(const float* __restrict__ X,
                                               const float* __restrict__ W, int ldw,
                                               float* __restrict__ Y, int ldy,
                                               int M, int mtile, int ntile, int lane) {
    const int half = lane >> 4;
    const int l    = lane & 15;

    int row  = mtile * 16 + l;
    int rowc = (row < M) ? row : (M - 1);   // clamp: keep all lanes on valid addrs
    const int col = ntile * 16 + l;
    const float* xrow = X + (size_t)rowc * K;

    v8f c = {};
#pragma unroll
    for (int k0 = 0; k0 < K; k0 += 32) {
        // A fragment: element e -> (M=l, K = k0 + 16*(e>>3) + 8*half + (e&7))
        const float* arow = xrow + k0 + 8 * half;
        float4 a0 = *(const float4*)(arow + 0);
        float4 a1 = *(const float4*)(arow + 4);
        float4 a2 = *(const float4*)(arow + 16);
        float4 a3 = *(const float4*)(arow + 20);
        v16h a;
        a[0]  = (_Float16)a0.x; a[1]  = (_Float16)a0.y;
        a[2]  = (_Float16)a0.z; a[3]  = (_Float16)a0.w;
        a[4]  = (_Float16)a1.x; a[5]  = (_Float16)a1.y;
        a[6]  = (_Float16)a1.z; a[7]  = (_Float16)a1.w;
        a[8]  = (_Float16)a2.x; a[9]  = (_Float16)a2.y;
        a[10] = (_Float16)a2.z; a[11] = (_Float16)a2.w;
        a[12] = (_Float16)a3.x; a[13] = (_Float16)a3.y;
        a[14] = (_Float16)a3.z; a[15] = (_Float16)a3.w;

        // B fragment: element e -> (K = k0 + 16*(e>>3) + 8*half + (e&7), N = l)
        v16h b;
        const float* wp = W + (size_t)(k0 + 8 * half) * ldw + col;
#pragma unroll
        for (int e = 0; e < 16; ++e)
            b[e] = (_Float16)wp[(size_t)((e & 7) + 16 * (e >> 3)) * ldw];

        c = __builtin_amdgcn_wmma_f32_16x16x32_f16(
                /*neg_a=*/false, a, /*neg_b=*/false, b,
                /*c_mod=*/(short)0, c, /*reuse_a=*/false, /*reuse_b=*/false);
    }

    // C/D layout: VGPR r -> M = mtile*16 + r + 8*half, N = col
    int m0 = mtile * 16 + 8 * half;
    if (mtile * 16 + 16 <= M) {
        // full tile: wave-uniform branch, unpredicated strided stores
        float* yp = Y + (size_t)m0 * ldy + col;
#pragma unroll
        for (int r = 0; r < 8; ++r)
            yp[(size_t)r * ldy] = c[r];
    } else {
#pragma unroll
        for (int r = 0; r < 8; ++r) {
            int mr = m0 + r;
            if (mr < M) Y[(size_t)mr * ldy + col] = c[r];
        }
    }
}

template <int K>
__global__ void gemm_wmma_f16_k(const float* __restrict__ X,
                                const float* __restrict__ W, int ldw,
                                float* __restrict__ Y, int ldy, int M) {
    gemm_tile_body<K>(X, W, ldw, Y, ldy, M, blockIdx.x, blockIdx.y,
                      threadIdx.x & 31);
}

// Generic fallback (runtime K, multiple of 32) in case dims differ.
__global__ void gemm_wmma_f16_any(const float* __restrict__ X, int ldx,
                                  const float* __restrict__ W, int ldw,
                                  float* __restrict__ Y, int ldy,
                                  int M, int K) {
    const int mtile = blockIdx.x, ntile = blockIdx.y;
    const int lane = threadIdx.x & 31, half = lane >> 4, l = lane & 15;
    int row  = mtile * 16 + l;
    int rowc = (row < M) ? row : (M - 1);
    const int col = ntile * 16 + l;
    v8f c = {};
    for (int k0 = 0; k0 < K; k0 += 32) {
        const float* arow = X + (size_t)rowc * ldx + k0 + 8 * half;
        float4 a0 = *(const float4*)(arow + 0);
        float4 a1 = *(const float4*)(arow + 4);
        float4 a2 = *(const float4*)(arow + 16);
        float4 a3 = *(const float4*)(arow + 20);
        v16h a;
        a[0]  = (_Float16)a0.x; a[1]  = (_Float16)a0.y;
        a[2]  = (_Float16)a0.z; a[3]  = (_Float16)a0.w;
        a[4]  = (_Float16)a1.x; a[5]  = (_Float16)a1.y;
        a[6]  = (_Float16)a1.z; a[7]  = (_Float16)a1.w;
        a[8]  = (_Float16)a2.x; a[9]  = (_Float16)a2.y;
        a[10] = (_Float16)a2.z; a[11] = (_Float16)a2.w;
        a[12] = (_Float16)a3.x; a[13] = (_Float16)a3.y;
        a[14] = (_Float16)a3.z; a[15] = (_Float16)a3.w;
        v16h b;
        const float* wp = W + (size_t)(k0 + 8 * half) * ldw + col;
#pragma unroll
        for (int e = 0; e < 16; ++e)
            b[e] = (_Float16)wp[(size_t)((e & 7) + 16 * (e >> 3)) * ldw];
        c = __builtin_amdgcn_wmma_f32_16x16x32_f16(false, a, false, b,
                                                   (short)0, c, false, false);
    }
    int m0 = mtile * 16 + 8 * half;
    if (mtile * 16 + 16 <= M) {
        float* yp = Y + (size_t)m0 * ldy + col;
#pragma unroll
        for (int r = 0; r < 8; ++r) yp[(size_t)r * ldy] = c[r];
    } else {
#pragma unroll
        for (int r = 0; r < 8; ++r) {
            int mr = m0 + r;
            if (mr < M) Y[(size_t)mr * ldy + col] = c[r];
        }
    }
}

// ---------------------------------------------------------------------------
// Propagation: out = D^-1/2 (A + I) D^-1/2 * g, then bias (+ReLU).
// mode 0: out is row-major [N x F].
// mode 1: F = 2*L features split: f <  L -> out[i*L+f]        (mu)
//                                 f >= L -> out[N*L + i*L+f-L] (logvar)
// ---------------------------------------------------------------------------
__device__ __forceinline__ size_t out_index(int i, int f, int mode, int N, int F) {
    if (mode == 0) return (size_t)i * F + f;
    int L = F >> 1;
    return (f < L) ? ((size_t)i * L + f)
                   : ((size_t)N * L + (size_t)i * L + (f - L));
}

__global__ void prop_init_kernel(const float* __restrict__ g,
                                 const float* __restrict__ dinv,
                                 float* __restrict__ out,
                                 int N, int F, int mode) {
    int idx = blockIdx.x * blockDim.x + threadIdx.x;
    if (idx >= N * F) return;
    int i = idx / F, f = idx - i * F;
    float d = dinv[i];
    out[out_index(i, f, mode, N, F)] = g[idx] * d * d;  // self-loop term
}

__global__ void prop_scatter_kernel(const float* __restrict__ g,
                                    const float* __restrict__ dinv,
                                    const int* __restrict__ ei,
                                    const float* __restrict__ ew,
                                    float* __restrict__ out,
                                    int N, int E, int F, int mode) {
    int chunks = F >> 2;  // float4 chunks per edge
    long long t = (long long)blockIdx.x * blockDim.x + threadIdx.x;
    if (t >= (long long)E * chunks) return;
    int e  = (int)(t / chunks);
    int f0 = (int)(t - (long long)e * chunks) << 2;
    int s = ei[e];
    int d = ei[(size_t)E + e];
    float norm = dinv[s] * ew[e] * dinv[d];
    float4 gv = *(const float4*)(g + (size_t)s * F + f0);
    float* o0 = &out[out_index(d, f0, mode, N, F)];  // 4 consecutive in both modes
    atomicAdd(o0 + 0, gv.x * norm);
    atomicAdd(o0 + 1, gv.y * norm);
    atomicAdd(o0 + 2, gv.z * norm);
    atomicAdd(o0 + 3, gv.w * norm);
}

__global__ void prop_finalize_kernel(float* __restrict__ out,
                                     const float* __restrict__ bias0,
                                     const float* __restrict__ bias1,
                                     int N, int F, int mode, int do_relu) {
    int idx = blockIdx.x * blockDim.x + threadIdx.x;
    if (idx >= N * F) return;
    int i = idx / F, f = idx - i * F;
    int L = F >> 1;
    float b = (mode == 1) ? ((f < L) ? bias0[f] : bias1[f - L]) : bias0[f];
    size_t o = out_index(i, f, mode, N, F);
    float v = out[o] + b;
    if (do_relu) v = fmaxf(v, 0.0f);
    out[o] = v;
}

// ---------------------------------------------------------------------------
// Launch
// ---------------------------------------------------------------------------
extern "C" void kernel_launch(void* const* d_in, const int* in_sizes, int n_in,
                              void* d_out, int out_size, void* d_ws, size_t ws_size,
                              hipStream_t stream) {
    const float* x   = (const float*)d_in[0];
    const int*   ei  = (const int*)d_in[1];   // [2, E] edge_index
    const float* ew  = (const float*)d_in[2];
    const float* W1  = (const float*)d_in[3];
    const float* b1  = (const float*)d_in[4];
    const float* Wmu = (const float*)d_in[5];
    const float* bmu = (const float*)d_in[6];
    const float* Wlv = (const float*)d_in[7];
    const float* blv = (const float*)d_in[8];

    const int hidden = in_sizes[4];             // 64
    const int latent = in_sizes[6];             // 32
    const int in_dim = in_sizes[3] / hidden;    // 128
    const int N      = in_sizes[0] / in_dim;    // 100000
    const int E      = in_sizes[2];             // 1600000
    const int F2     = 2 * latent;              // 64 (= hidden here)

    // Workspace: dinv [N], bufA [N*hidden]. d_out (N*2*latent == N*hidden floats)
    // doubles as the layer-1 accumulator / h1 buffer.
    float* ws   = (float*)d_ws;
    float* dinv = ws;
    size_t off  = ((size_t)N + 63) & ~(size_t)63;
    float* bufA = ws + off;
    float* out  = (float*)d_out;

    const int T = 256;
    dim3 blk(T);

    auto gemm = [&](const float* X, const float* W, int ldw, float* Y, int ldy,
                    int M, int K, int Ncols) {
        dim3 grid((M + 15) / 16, Ncols / 16);
        if (K == 128)
            gemm_wmma_f16_k<128><<<grid, dim3(32), 0, stream>>>(X, W, ldw, Y, ldy, M);
        else if (K == 64)
            gemm_wmma_f16_k<64><<<grid, dim3(32), 0, stream>>>(X, W, ldw, Y, ldy, M);
        else
            gemm_wmma_f16_any<<<grid, dim3(32), 0, stream>>>(X, K, W, ldw, Y, ldy, M, K);
    };

    // 1) degrees -> dinv
    deg_init_kernel<<<dim3((N + T - 1) / T), blk, 0, stream>>>(dinv, N);
    deg_scatter_kernel<<<dim3((E + T - 1) / T), blk, 0, stream>>>(dinv, ei, ew, E);
    deg_rsqrt_kernel<<<dim3((N + T - 1) / T), blk, 0, stream>>>(dinv, N);

    // 2) h0 = x @ W1  -> bufA [N x hidden]
    gemm(x, W1, hidden, bufA, hidden, N, in_dim, hidden);

    // 3) propagate layer 1 into d_out, then h1 = relu(acc + b1) in place
    {
        int tot = N * hidden;
        prop_init_kernel<<<dim3((tot + T - 1) / T), blk, 0, stream>>>(
            bufA, dinv, out, N, hidden, 0);
        long long st = (long long)E * (hidden >> 2);
        prop_scatter_kernel<<<dim3((unsigned)((st + T - 1) / T)), blk, 0, stream>>>(
            bufA, dinv, ei, ew, out, N, E, hidden, 0);
        prop_finalize_kernel<<<dim3((tot + T - 1) / T), blk, 0, stream>>>(
            out, b1, nullptr, N, hidden, 0, 1);
    }

    // 4) g = [h1@Wmu | h1@Wlv] -> bufA [N x 2*latent]
    gemm(out, Wmu, latent, bufA,          F2, N, hidden, latent);
    gemm(out, Wlv, latent, bufA + latent, F2, N, hidden, latent);

    // 5) propagate layer 2, split-accumulating straight into (mu | logvar)
    {
        int tot = N * F2;
        prop_init_kernel<<<dim3((tot + T - 1) / T), blk, 0, stream>>>(
            bufA, dinv, out, N, F2, 1);
        long long st = (long long)E * (F2 >> 2);
        prop_scatter_kernel<<<dim3((unsigned)((st + T - 1) / T)), blk, 0, stream>>>(
            bufA, dinv, ei, ew, out, N, E, F2, 1);
        prop_finalize_kernel<<<dim3((tot + T - 1) / T), blk, 0, stream>>>(
            out, bmu, blv, N, F2, 1, 0);
    }
}
